// SMCCU_71296457113751
// MI455X (gfx1250) — matrix operand order repaired and checked by reading, not compiled, not run on hardware
//
#include <hip/hip_runtime.h>

// ------------------------------------------------------------------
// Sizes (from the reference): B=8, DIM=256, H=W=64, D_INNER=512,
// NHEADS=8, HEADDIM=64, D_STATE=64, CONV_DIM=640, D_IN_PROJ=1160.
// ------------------------------------------------------------------
#define NPIX     32768      // B*H*W
#define CH       256
#define DINNER   512
#define NHEADS_  8
#define CONVDIM_ 640
#define NPROJ_   1160
#define NPROJP_  1216       // padded to a multiple of 64 for the WMMA tiler
#define CONVK_   2304       // 9 * 256 implicit-GEMM K for the 3x3 conv

typedef __attribute__((ext_vector_type(16))) __bf16       v16bf;
typedef __attribute__((ext_vector_type(8)))  float        v8f;
typedef __attribute__((ext_vector_type(4)))  unsigned int uint4e;
typedef int v4i_gnu __attribute__((vector_size(16)));   // matches async builtin param

#define AS1_ __attribute__((address_space(1)))
#define AS3_ __attribute__((address_space(3)))

// gfx1250 async global->LDS data mover (ASYNCcnt-tracked), with a
// compile-safe fallback if this toolchain doesn't expose the builtins.
#if defined(__has_builtin)
#  if __has_builtin(__builtin_amdgcn_global_load_async_to_lds_b128) && \
      __has_builtin(__builtin_amdgcn_s_wait_asynccnt)
#    define USE_ASYNC_LDS 1
#  endif
#endif

union Frag { uint4e u[2]; v16bf v; };

// A-matrix fragment (16x32 bf16): lane holds row M=lane&15,
// K range [kk, kk+16) with kk = k0 + 16*(lane>=16).
__device__ __forceinline__ v16bf load_a_frag(const __bf16* rowPtr, int k0, int lane) {
  Frag f;
  int kk = k0 + ((lane >> 4) << 4);
  f.u[0] = *(const uint4e*)(rowPtr + kk);
  f.u[1] = *(const uint4e*)(rowPtr + kk + 8);
  return f.v;
}
// B-matrix fragment (32x16 bf16) from an LDS tile laid out [n_local][32]:
// lane holds col N=lane&15, K range [kb, kb+16) with kb = 16*(lane>=16).
__device__ __forceinline__ v16bf load_b_frag_lds(const __bf16* tile, int lane) {
  Frag f;
  const __bf16* p = tile + ((lane >> 4) << 4);
  f.u[0] = *(const uint4e*)(p);
  f.u[1] = *(const uint4e*)(p + 8);
  return f.v;
}
// B fragment straight from global weights W[N][K] row-major.
__device__ __forceinline__ v16bf load_b_frag(const __bf16* rowPtr, int k0, int lane) {
  Frag f;
  int kb = k0 + ((lane >> 4) << 4);
  f.u[0] = *(const uint4e*)(rowPtr + kb);
  f.u[1] = *(const uint4e*)(rowPtr + kb + 8);
  return f.v;
}

__device__ __forceinline__ float sigmoidf_(float v) { return 1.0f / (1.0f + expf(-v)); }
__device__ __forceinline__ float siluf_(float v)    { return v / (1.0f + expf(-v)); }
__device__ __forceinline__ float geluf_(float v)    { return 0.5f * v * (1.0f + erff(v * 0.70710678118654752f)); }

// ------------------------------------------------------------------
// Generic bf16 WMMA GEMM:  C[M,N] = act(A[M,K] @ W[N,K]^T + bias)
// A packed lda=K, W packed ldw=K. Block = 8 waves, each wave = 16x64 tile;
// the 64x32 B tile for the whole block is staged in LDS via the async
// data mover (all 8 waves share it), A rows stream from global.
// grid = (M/128, N/64). Store f32 (Cf) and/or bf16 (Cb) with stride ldc.
// ------------------------------------------------------------------
__global__ void __launch_bounds__(256) gemm_bf16_wmma(
    const __bf16* __restrict__ A, const __bf16* __restrict__ W,
    const float* __restrict__ bias, float* __restrict__ Cf, __bf16* __restrict__ Cb,
    int K, int ldc, int act)
{
  __shared__ __align__(16) __bf16 ldsB[64 * 32];   // 4 KB B tile: [n_local][k_local]

  const int t    = threadIdx.x;
  const int lane = t & 31;
  const int wave = t >> 5;
  const int mt   = blockIdx.x * 8 + wave;
  const int nb   = blockIdx.y * 64;
  const int rowA = mt * 16 + (lane & 15);

  // staging map: thread t copies 16 bytes of row nb+(t>>2), chunk t&3
  const int sn  = t >> 2;
  const int sch = t & 3;
  const __bf16* gB = W + (size_t)(nb + sn) * K + sch * 8;
  __bf16*       lB = ldsB + sn * 32 + sch * 8;

  v8f acc[4] = {};
  const __bf16* aRow = A + (size_t)rowA * K;

  for (int k0 = 0; k0 < K; k0 += 32) {
#ifdef USE_ASYNC_LDS
    __builtin_amdgcn_global_load_async_to_lds_b128(
        (AS1_ v4i_gnu*)(const void*)(gB + k0),
        (AS3_ v4i_gnu*)(void*)lB, 0, 0);
    __builtin_amdgcn_s_wait_asynccnt(0);
#else
    *(uint4e*)lB = *(const uint4e*)(gB + k0);
#endif
    __syncthreads();

    __builtin_prefetch(aRow + k0 + 128, 0, 1);      // global_prefetch_b8
    v16bf af = load_a_frag(aRow, k0, lane);
#pragma unroll
    for (int j = 0; j < 4; ++j) {
      v16bf bfr = load_b_frag_lds(ldsB + (j * 16 + (lane & 15)) * 32, lane);
      acc[j] = __builtin_amdgcn_wmma_f32_16x16x32_bf16(
          false, af, false, bfr, (short)0, acc[j], false, false);
    }
    __syncthreads();
  }

  const int rb = mt * 16 + ((lane < 16) ? 0 : 8);
  const int cl = lane & 15;
#pragma unroll
  for (int j = 0; j < 4; ++j) {
    int col = nb + j * 16 + cl;
    float bv = bias ? bias[col] : 0.0f;
#pragma unroll
    for (int e = 0; e < 8; ++e) {
      float v = acc[j][e] + bv;
      if (act == 1) v = sigmoidf_(v);
      size_t idx = (size_t)(rb + e) * ldc + col;
      if (Cf) Cf[idx] = v;
      if (Cb) Cb[idx] = (__bf16)v;
    }
  }
}

// ------------------------------------------------------------------
// 3x3 conv (NHWC, 256->256) as implicit GEMM: 9 shifted taps accumulated
// in WMMA registers, fused BN + exact GELU epilogue.
// grid = (NPIX/128, 256/64), block 256.
// ------------------------------------------------------------------
__global__ void __launch_bounds__(256) conv3x3_wmma_kernel(
    const __bf16* __restrict__ xn, const __bf16* __restrict__ Wc,
    const float* __restrict__ g, const float* __restrict__ bb,
    const float* __restrict__ mm, const float* __restrict__ vv,
    float* __restrict__ lf32, __bf16* __restrict__ lf16)
{
  const int lane = threadIdx.x & 31;
  const int wave = threadIdx.x >> 5;
  const int mt   = blockIdx.x * 8 + wave;
  const int nb   = blockIdx.y * 64;

  const int pix  = mt * 16 + (lane & 15);
  const int bi   = pix >> 12, yy = (pix >> 6) & 63, xx = pix & 63;

  v8f acc[4] = {};
  const __bf16* wBase = Wc + (size_t)(nb + (lane & 15)) * CONVK_;

  for (int tap = 0; tap < 9; ++tap) {
    int sy = yy + tap / 3 - 1;
    int sx = xx + tap % 3 - 1;
    bool valid = ((unsigned)sy < 64u) && ((unsigned)sx < 64u);
    const __bf16* aRow = xn + ((size_t)((bi << 12) | (sy << 6) | sx)) * CH;
    for (int k0 = 0; k0 < CH; k0 += 32) {
      v16bf af;
      if (valid) af = load_a_frag(aRow, k0, lane);
      else { Frag z; z.u[0] = (uint4e){0u,0u,0u,0u}; z.u[1] = (uint4e){0u,0u,0u,0u}; af = z.v; }
#pragma unroll
      for (int j = 0; j < 4; ++j) {
        v16bf bfr = load_b_frag(wBase + (size_t)j * 16 * CONVK_ + tap * CH, k0, lane);
        acc[j] = __builtin_amdgcn_wmma_f32_16x16x32_bf16(
            false, af, false, bfr, (short)0, acc[j], false, false);
      }
    }
  }

  const int rb = mt * 16 + ((lane < 16) ? 0 : 8);
  const int cl = lane & 15;
#pragma unroll
  for (int j = 0; j < 4; ++j) {
    int col = nb + j * 16 + cl;
    float sc = rsqrtf(vv[col] + 1e-5f) * g[col];
#pragma unroll
    for (int e = 0; e < 8; ++e) {
      float v  = (acc[j][e] - mm[col]) * sc + bb[col];
      float ge = geluf_(v);
      size_t idx = (size_t)(rb + e) * CH + col;
      lf32[idx] = ge;
      lf16[idx] = (__bf16)ge;
    }
  }
}

// ---------------- pre-LN over channels of NCHW input -> NHWC bf16 ---------
__global__ void __launch_bounds__(256) preln_kernel(
    const float* __restrict__ x, const float* __restrict__ w,
    const float* __restrict__ b, __bf16* __restrict__ dst)
{
  int p = blockIdx.x, t = threadIdx.x;
  int bi = p >> 12, yy = (p >> 6) & 63, xx = p & 63;
  float v = x[(((size_t)(bi * CH + t)) << 12) | (yy << 6) | xx];
  __shared__ float s1[256], s2[256];
  s1[t] = v; s2[t] = v * v; __syncthreads();
  for (int o = 128; o > 0; o >>= 1) {
    if (t < o) { s1[t] += s1[t + o]; s2[t] += s2[t + o]; }
    __syncthreads();
  }
  float mean = s1[0] * (1.0f / 256.0f);
  float var  = s2[0] * (1.0f / 256.0f) - mean * mean;
  float r    = rsqrtf(var + 1e-6f);
  dst[(size_t)p * CH + t] = (__bf16)((v - mean) * r * w[t] + b[t]);
}

// ------- LN over last dim of NHWC f32 -> bf16; order=1 gathers transposed --
__global__ void __launch_bounds__(256) ln_nhwc_kernel(
    const float* __restrict__ src, const float* __restrict__ w,
    const float* __restrict__ b, __bf16* __restrict__ dst, int order)
{
  int pp = blockIdx.x, t = threadIdx.x;
  int sp = pp;
  if (order) {           // sequence order (b, w, h) -> source pixel (b, h, w)
    int bi = pp >> 12, wv = (pp >> 6) & 63, hv = pp & 63;
    sp = (bi << 12) | (hv << 6) | wv;
  }
  float v = src[(size_t)sp * CH + t];
  __shared__ float s1[256], s2[256];
  s1[t] = v; s2[t] = v * v; __syncthreads();
  for (int o = 128; o > 0; o >>= 1) {
    if (t < o) { s1[t] += s1[t + o]; s2[t] += s2[t + o]; }
    __syncthreads();
  }
  float mean = s1[0] * (1.0f / 256.0f);
  float var  = s2[0] * (1.0f / 256.0f) - mean * mean;
  float r    = rsqrtf(var + 1e-5f);
  dst[(size_t)pp * CH + t] = (__bf16)((v - mean) * r * w[t] + b[t]);
}

// ------------- dt = softplus(zx[:,1152+h] + dt_bias); decay = exp(dt*A) ----
__global__ void dt_decay_kernel(const float* __restrict__ zx,
    const float* __restrict__ dt_bias, const float* __restrict__ A_log,
    float* __restrict__ dtb, float* __restrict__ dec)
{
  int tid = blockIdx.x * 256 + threadIdx.x;    // NPIX*8
  int p = tid >> 3, h = tid & 7;
  float raw = zx[(size_t)p * NPROJP_ + DINNER + CONVDIM_ + h] + dt_bias[h];
  float dt  = (raw > 20.0f) ? raw : log1pf(expf(raw));
  dtb[tid] = dt;
  dec[tid] = expf(dt * (-expf(A_log[h])));
}

// ---------- depthwise causal conv1d (width 4) + SiLU over xBC -------------
__global__ void conv1d_silu_kernel(const float* __restrict__ zx,
    const float* __restrict__ cw, const float* __restrict__ cb,
    float* __restrict__ xbc)
{
  size_t tid = (size_t)blockIdx.x * 256 + threadIdx.x;   // NPIX*640
  int c = (int)(tid % CONVDIM_);
  int p = (int)(tid / CONVDIM_);
  int n = p >> 6, l = p & 63;
  float acc = cb[c];
#pragma unroll
  for (int k = 0; k < 4; ++k) {
    int ll = l - 3 + k;
    if (ll >= 0) acc += cw[c * 4 + k] * zx[(size_t)((n << 6) + ll) * NPROJP_ + DINNER + c];
  }
  xbc[tid] = siluf_(acc);
}

// ---------- selective-scan: one block per (sequence, head) ----------------
// State 64x64 lives in registers: thread t owns p=t>>2, s in [16*(t&3),+16).
__global__ void __launch_bounds__(256) ssm_scan_kernel(
    const float* __restrict__ xbc, const float* __restrict__ dtb,
    const float* __restrict__ dec, float* __restrict__ y32)
{
  const int nseq = blockIdx.x >> 3;
  const int head = blockIdx.x & 7;
  const int t = threadIdx.x, j = t & 3, p = t >> 2;

  __shared__ float sB[64], sC[64], sX[64], sSc[2];
  float st[16];
#pragma unroll
  for (int i = 0; i < 16; ++i) st[i] = 0.0f;

  for (int l = 0; l < 64; ++l) {
    int pix = (nseq << 6) + l;
    if (t < 64)        sB[t]       = xbc[(size_t)pix * CONVDIM_ + DINNER + t];
    else if (t < 128)  sC[t - 64]  = xbc[(size_t)pix * CONVDIM_ + DINNER + 64 + (t - 64)];
    else if (t < 192)  sX[t - 128] = xbc[(size_t)pix * CONVDIM_ + head * 64 + (t - 128)];
    else if (t == 192) sSc[0]      = dtb[pix * 8 + head];
    else if (t == 193) sSc[1]      = dec[pix * 8 + head];
    __syncthreads();

    float dtx = sSc[0] * sX[p];
    float dcy = sSc[1];
    float ys  = 0.0f;
#pragma unroll
    for (int i = 0; i < 16; ++i) {
      int s = j * 16 + i;
      st[i] = dcy * st[i] + dtx * sB[s];
      ys += st[i] * sC[s];
    }
    ys += __shfl_xor(ys, 1);
    ys += __shfl_xor(ys, 2);
    if (j == 0) y32[(size_t)pix * DINNER + head * 64 + p] = ys;
    __syncthreads();
  }
}

// ---- y = (ys + D*x) * silu(z); RMS-norm over 512; * norm_w -> bf16 -------
__global__ void __launch_bounds__(256) rmsgate_kernel(
    const float* __restrict__ y32, const float* __restrict__ xbc,
    const float* __restrict__ zx, const float* __restrict__ Dp,
    const float* __restrict__ nw, __bf16* __restrict__ ybf)
{
  int p = blockIdx.x, t = threadIdx.x;
  float v[2];
#pragma unroll
  for (int i = 0; i < 2; ++i) {
    int c = t + i * 256;
    float yv = y32[(size_t)p * DINNER + c] + Dp[c >> 6] * xbc[(size_t)p * CONVDIM_ + c];
    float z  = zx[(size_t)p * NPROJP_ + c];
    v[i] = yv * siluf_(z);
  }
  __shared__ float s1[256];
  s1[t] = v[0] * v[0] + v[1] * v[1]; __syncthreads();
  for (int o = 128; o > 0; o >>= 1) {
    if (t < o) s1[t] += s1[t + o];
    __syncthreads();
  }
  float scale = rsqrtf(s1[0] * (1.0f / 512.0f) + 1e-5f);
#pragma unroll
  for (int i = 0; i < 2; ++i) {
    int c = t + i * 256;
    ybf[(size_t)p * DINNER + c] = (__bf16)(v[i] * scale * nw[c]);
  }
}

// ---- scatter col-mamba output (sequence order) into rowcol[:,256:] -------
__global__ void colout_transpose_kernel(const __bf16* __restrict__ cp,
                                        __bf16* __restrict__ rc)
{
  int tid = blockIdx.x * 256 + threadIdx.x;   // NPIX*256
  int pp = tid >> 8, c = tid & 255;
  int bi = pp >> 12, wv = (pp >> 6) & 63, hv = pp & 63;
  int p = (bi << 12) | (hv << 6) | wv;
  rc[(size_t)p * DINNER + CH + c] = cp[(size_t)pp * CH + c];
}

// ---- seq = g*row + (1-g)*col ; build fuse input [lf | seq] in bf16 -------
__global__ void seq_mix_kernel(const __bf16* __restrict__ rc,
    const float* __restrict__ gbuf, const __bf16* __restrict__ lf16,
    __bf16* __restrict__ fa)
{
  int tid = blockIdx.x * 256 + threadIdx.x;   // NPIX*256
  int p = tid >> 8, c = tid & 255;
  float r  = (float)rc[(size_t)p * DINNER + c];
  float co = (float)rc[(size_t)p * DINNER + CH + c];
  float g  = gbuf[(size_t)p * CH + c];
  fa[(size_t)p * DINNER + c]      = lf16[(size_t)p * CH + c];
  fa[(size_t)p * DINNER + CH + c] = (__bf16)(g * r + (1.0f - g) * co);
}

// ---- out = x + gamma * gelu(bn(f)) back to NCHW --------------------------
__global__ void final_kernel(const float* __restrict__ x, const float* __restrict__ f,
    const float* __restrict__ gamma, const float* __restrict__ g,
    const float* __restrict__ bb, const float* __restrict__ mm,
    const float* __restrict__ vv, float* __restrict__ out)
{
  size_t tid = (size_t)blockIdx.x * 256 + threadIdx.x;  // 8*256*64*64
  int c  = (int)((tid >> 12) & 255);
  int yy = (int)((tid >> 6) & 63), xx = (int)(tid & 63);
  int bi = (int)(tid >> 20);
  size_t p = ((size_t)bi << 12) | (yy << 6) | xx;
  float fv = f[p * CH + c];
  float bn = (fv - mm[c]) * rsqrtf(vv[c] + 1e-5f) * g[c] + bb[c];
  out[tid] = x[tid] + gamma[c] * geluf_(bn);
}

// --------------------------- weight prep ----------------------------------
__global__ void cvt_bf16_kernel(const float* __restrict__ s, __bf16* __restrict__ d, int n) {
  int tid = blockIdx.x * 256 + threadIdx.x;
  if (tid < n) d[tid] = (__bf16)s[tid];
}
__global__ void pad_inproj_kernel(const float* __restrict__ s, __bf16* __restrict__ d) {
  int tid = blockIdx.x * 256 + threadIdx.x;          // NPROJP_*256
  d[tid] = (tid < NPROJ_ * CH) ? (__bf16)s[tid] : (__bf16)0.0f;
}
__global__ void convw_rearrange_kernel(const float* __restrict__ s, __bf16* __restrict__ d) {
  int tid = blockIdx.x * 256 + threadIdx.x;          // 256*2304
  int o = tid / CONVK_, r = tid % CONVK_;
  int tap = r >> 8, i = r & 255;
  int ky = tap / 3, kx = tap % 3;
  d[tid] = (__bf16)s[((o * CH + i) * 3 + ky) * 3 + kx];
}

// --------------------------------------------------------------------------
static void run_mamba_phase(hipStream_t stream, int order,
    const float* lf32,
    const float* ln_w, const float* ln_b,
    const __bf16* w_inproj, const float* conv_w, const float* conv_b,
    const float* dt_bias, const float* A_log, const float* Dp, const float* norm_w,
    const __bf16* w_outproj,
    __bf16* u, float* zx, float* xbc, float* dtb, float* dec,
    float* y32, __bf16* ybf, __bf16* outStore, int out_ldc)
{
  ln_nhwc_kernel<<<NPIX, 256, 0, stream>>>(lf32, ln_w, ln_b, u, order);
  gemm_bf16_wmma<<<dim3(NPIX / 128, NPROJP_ / 64), 256, 0, stream>>>(
      u, w_inproj, nullptr, zx, nullptr, CH, NPROJP_, 0);
  dt_decay_kernel<<<NPIX * 8 / 256, 256, 0, stream>>>(zx, dt_bias, A_log, dtb, dec);
  conv1d_silu_kernel<<<NPIX * CONVDIM_ / 256, 256, 0, stream>>>(zx, conv_w, conv_b, xbc);
  ssm_scan_kernel<<<(NPIX / 64) * NHEADS_, 256, 0, stream>>>(xbc, dtb, dec, y32);
  rmsgate_kernel<<<NPIX, 256, 0, stream>>>(y32, xbc, zx, Dp, norm_w, ybf);
  gemm_bf16_wmma<<<dim3(NPIX / 128, CH / 64), 256, 0, stream>>>(
      ybf, w_outproj, nullptr, nullptr, outStore, DINNER, out_ldc, 0);
}

extern "C" void kernel_launch(void* const* d_in, const int* in_sizes, int n_in,
                              void* d_out, int out_size, void* d_ws, size_t ws_size,
                              hipStream_t stream) {
  (void)in_sizes; (void)n_in; (void)out_size; (void)ws_size;
  // Inputs flattened in setup_inputs() dict order (nested dicts in insertion order):
  const float* x        = (const float*)d_in[0];
  const float* pre_w    = (const float*)d_in[1];
  const float* pre_b    = (const float*)d_in[2];
  const float* lconv_w  = (const float*)d_in[3];
  const float* bn_g     = (const float*)d_in[4];
  const float* bn_b     = (const float*)d_in[5];
  const float* bn_m     = (const float*)d_in[6];
  const float* bn_v     = (const float*)d_in[7];
  const float* row_ln_w = (const float*)d_in[8];
  const float* row_ln_b = (const float*)d_in[9];
  const float* col_ln_w = (const float*)d_in[10];
  const float* col_ln_b = (const float*)d_in[11];
  const float* r_inproj = (const float*)d_in[12];
  const float* r_convw  = (const float*)d_in[13];
  const float* r_convb  = (const float*)d_in[14];
  const float* r_dtb    = (const float*)d_in[15];
  const float* r_alog   = (const float*)d_in[16];
  const float* r_D      = (const float*)d_in[17];
  const float* r_normw  = (const float*)d_in[18];
  const float* r_outp   = (const float*)d_in[19];
  const float* c_inproj = (const float*)d_in[20];
  const float* c_convw  = (const float*)d_in[21];
  const float* c_convb  = (const float*)d_in[22];
  const float* c_dtb    = (const float*)d_in[23];
  const float* c_alog   = (const float*)d_in[24];
  const float* c_D      = (const float*)d_in[25];
  const float* c_normw  = (const float*)d_in[26];
  const float* c_outp   = (const float*)d_in[27];
  const float* gate_w   = (const float*)d_in[28];
  const float* gate_b   = (const float*)d_in[29];
  const float* fuse_w   = (const float*)d_in[30];
  const float* fbn_g    = (const float*)d_in[31];
  const float* fbn_b    = (const float*)d_in[32];
  const float* fbn_m    = (const float*)d_in[33];
  const float* fbn_v    = (const float*)d_in[34];
  const float* gamma    = (const float*)d_in[35];

  // ---- workspace carve-out (~500 MB) ----
  char* ws = (char*)d_ws;
  size_t off = 0;
  auto carve = [&](size_t bytes) -> void* {
    off = (off + 255) & ~(size_t)255;
    void* p = ws + off; off += bytes; return p;
  };
  float*  lf32 = (float*)carve((size_t)NPIX * CH * 4);
  __bf16* lf16 = (__bf16*)carve((size_t)NPIX * CH * 2);
  __bf16* actA = (__bf16*)carve((size_t)NPIX * CH * 2);          // xn, then u
  float*  zx   = (float*)carve((size_t)NPIX * NPROJP_ * 4);      // also reused for f
  float*  xbc  = (float*)carve((size_t)NPIX * CONVDIM_ * 4);
  float*  dtb  = (float*)carve((size_t)NPIX * 8 * 4);
  float*  dec  = (float*)carve((size_t)NPIX * 8 * 4);
  float*  y32  = (float*)carve((size_t)NPIX * DINNER * 4);       // also reused for g
  __bf16* ybf  = (__bf16*)carve((size_t)NPIX * DINNER * 2);
  __bf16* rc   = (__bf16*)carve((size_t)NPIX * DINNER * 2);      // [row | col]
  __bf16* cp   = (__bf16*)carve((size_t)NPIX * CH * 2);          // col out (seq order)
  __bf16* fa   = (__bf16*)carve((size_t)NPIX * DINNER * 2);      // [lf | seq]
  __bf16* wconv = (__bf16*)carve((size_t)CH * CONVK_ * 2);
  __bf16* winpR = (__bf16*)carve((size_t)NPROJP_ * CH * 2);
  __bf16* winpC = (__bf16*)carve((size_t)NPROJP_ * CH * 2);
  __bf16* woutR = (__bf16*)carve((size_t)CH * DINNER * 2);
  __bf16* woutC = (__bf16*)carve((size_t)CH * DINNER * 2);
  __bf16* wgate = (__bf16*)carve((size_t)CH * DINNER * 2);
  __bf16* wfuse = (__bf16*)carve((size_t)CH * DINNER * 2);

  // ---- weight prep (f32 -> bf16, layouts for WMMA B-fragments) ----
  convw_rearrange_kernel<<<CH * CONVK_ / 256, 256, 0, stream>>>(lconv_w, wconv);
  pad_inproj_kernel<<<NPROJP_ * CH / 256, 256, 0, stream>>>(r_inproj, winpR);
  pad_inproj_kernel<<<NPROJP_ * CH / 256, 256, 0, stream>>>(c_inproj, winpC);
  cvt_bf16_kernel<<<CH * DINNER / 256, 256, 0, stream>>>(r_outp, woutR, CH * DINNER);
  cvt_bf16_kernel<<<CH * DINNER / 256, 256, 0, stream>>>(c_outp, woutC, CH * DINNER);
  cvt_bf16_kernel<<<CH * DINNER / 256, 256, 0, stream>>>(gate_w, wgate, CH * DINNER);
  cvt_bf16_kernel<<<CH * DINNER / 256, 256, 0, stream>>>(fuse_w, wfuse, CH * DINNER);

  // ---- pre-LN + 3x3 conv + BN + GELU ----
  preln_kernel<<<NPIX, 256, 0, stream>>>(x, pre_w, pre_b, actA);
  conv3x3_wmma_kernel<<<dim3(NPIX / 128, CH / 64), 256, 0, stream>>>(
      actA, wconv, bn_g, bn_b, bn_m, bn_v, lf32, lf16);

  // ---- row mamba (sequences = image rows), write rc[:, :256] ----
  run_mamba_phase(stream, 0, lf32, row_ln_w, row_ln_b, winpR, r_convw, r_convb,
                  r_dtb, r_alog, r_D, r_normw, woutR,
                  actA, zx, xbc, dtb, dec, y32, ybf, rc, DINNER);

  // ---- col mamba (sequences = image cols, transposed order) ----
  run_mamba_phase(stream, 1, lf32, col_ln_w, col_ln_b, winpC, c_convw, c_convb,
                  c_dtb, c_alog, c_D, c_normw, woutC,
                  actA, zx, xbc, dtb, dec, y32, ybf, cp, CH);
  colout_transpose_kernel<<<NPIX * CH / 256, 256, 0, stream>>>(cp, rc);

  // ---- gate = sigmoid([row|col] @ gate_w^T + b); seq mix; fuse GEMM ----
  float* gbuf = y32;   // reuse
  gemm_bf16_wmma<<<dim3(NPIX / 128, CH / 64), 256, 0, stream>>>(
      rc, wgate, gate_b, gbuf, nullptr, DINNER, CH, 1);
  seq_mix_kernel<<<NPIX * CH / 256, 256, 0, stream>>>(rc, gbuf, lf16, fa);

  float* fbuf = zx;    // reuse
  gemm_bf16_wmma<<<dim3(NPIX / 128, CH / 64), 256, 0, stream>>>(
      fa, wfuse, nullptr, fbuf, nullptr, DINNER, CH, 0);

  // ---- residual + gamma * gelu(bn(f)) in NCHW ----
  final_kernel<<<(int)(((size_t)NPIX * CH) / 256), 256, 0, stream>>>(
      x, fbuf, gamma, fbn_g, fbn_b, fbn_m, fbn_v, (float*)d_out);
}
